// ContralateralAttention_55722905698844
// MI455X (gfx1250) — compile-verified, matrix operand order checked by reading, not run on hardware
//
#include <hip/hip_runtime.h>
#include <hip/hip_bf16.h>

typedef __attribute__((ext_vector_type(16))) _Float16 v16h;
typedef __attribute__((ext_vector_type(8)))  _Float16 v8h;
typedef __attribute__((ext_vector_type(8)))  float    v8f;

#define NBS    16          // 8 batches * 2 sides
#define NPIX   2048        // 64 * 32 pixels per half
#define CDIM   256
#define HDIM   64
#define LNP    264         // LDS pitch (floats) for LN tile
#define APITCH 1032        // LDS pitch (halves) for asym tile: 1024 + 8 pad

// ---------------------------------------------------------------------------
// WMMA fragment loader (global): row-major matrix, K-contiguous, ld = K-stride.
// Per CDNA5 ISA 7.12.2 (16-bit A 16x32): lanes 0-15 -> row=lane, K {0..7,16..23};
// lanes 16-31 -> row=lane-16, K {8..15,24..31}. B (=W^T) mirrors A's layout.
// ---------------------------------------------------------------------------
__device__ __forceinline__ v16h load_frag_k(const _Float16* __restrict__ base,
                                            int ld, int row0, int k0, int lane) {
  const int l  = lane & 15;
  const int hi = lane >> 4;
  const _Float16* p = base + (size_t)(row0 + l) * ld + (k0 + hi * 8);
  union { v16h v; v8h h[2]; } u;
  u.h[0] = *(const v8h*)(p);
  u.h[1] = *(const v8h*)(p + 16);
  return u.v;
}

// Same fragment pattern out of an LDS tile (row pitch in halves).
__device__ __forceinline__ v16h load_frag_lds(const _Float16* base,
                                              int pitch, int k0, int lane) {
  const int l  = lane & 15;
  const int hi = lane >> 4;
  const _Float16* p = base + l * pitch + k0 + hi * 8;
  union { v16h v; v8h h[2]; } u;
  u.h[0] = *(const v8h*)(p);
  u.h[1] = *(const v8h*)(p + 16);
  return u.v;
}

// ---------------------------------------------------------------------------
// Elementwise f32 -> f16 (weights)
// ---------------------------------------------------------------------------
__global__ void f32_to_f16(const float* __restrict__ in, _Float16* __restrict__ out, int n) {
  int i = blockIdx.x * blockDim.x + threadIdx.x;
  if (i < n) out[i] = (_Float16)in[i];
}

// ---------------------------------------------------------------------------
// Build Xq (own half) and Xm (mirrored opposite half) as (NBS, NPIX, CDIM) f16
// feat: (8, 256, 64, 64) f32.  p = y*32 + x.
//   Xq[b,s,p,c] = feat[b,c,y, s*32 + x]
//   Xm[b,s,p,c] = feat[b,c,y, (1-s)*32 + (31-x)]   (mirror of opposite half)
// ---------------------------------------------------------------------------
__global__ void prep_inputs(const float* __restrict__ feat,
                            _Float16* __restrict__ Xq,
                            _Float16* __restrict__ Xm) {
  int idx = blockIdx.x * blockDim.x + threadIdx.x;   // NBS*CDIM*NPIX threads
  if (idx >= NBS * CDIM * NPIX) return;
  int p  = idx & (NPIX - 1);
  int c  = (idx >> 11) & (CDIM - 1);
  int bs = idx >> 19;
  int b = bs >> 1, s = bs & 1;
  int y = p >> 5, x = p & 31;
  const float* f = feat + (((size_t)b * CDIM + c) * 64 + y) * 64;
  size_t o = ((size_t)bs * NPIX + p) * CDIM + c;
  Xq[o] = (_Float16)f[s * 32 + x];
  Xm[o] = (_Float16)f[(1 - s) * 32 + (31 - x)];
}

// ---------------------------------------------------------------------------
// Batched GEMM: Out[r, m] = sum_k A[r,k] * W[m,k] + bias[m]   (f16 in/out, f32 acc)
// grid: (rows/64, M/64, NBS), block: 128 (4 waves). Wave computes a 16x64 tile.
// Ping-pong double buffering, straight-line body (last double-step peeled so
// the refill is unconditional): no shift copies, no spills. K % 64 == 0.
// ---------------------------------------------------------------------------
template<int K>
__global__ void gemm_xwt_f16(const _Float16* __restrict__ A, size_t aBatchStride,
                             const _Float16* __restrict__ W,
                             const float*    __restrict__ bias,
                             _Float16* __restrict__ Out, size_t oBatchStride,
                             int M) {
  const int lane = threadIdx.x & 31;
  const int wave = threadIdx.x >> 5;
  const int batch = blockIdx.z;
  const int row0 = blockIdx.x * 64 + wave * 16;
  const int colBase = blockIdx.y * 64;
  const _Float16* Ab = A + (size_t)batch * aBatchStride;

  v16h A0 = load_frag_k(Ab, K, row0, 0, lane);
  v16h B0[4];
#pragma unroll
  for (int t = 0; t < 4; ++t)
    B0[t] = load_frag_k(W, K, colBase + t * 16, 0, lane);

  v8f acc[4] = {};
#pragma unroll 1
  for (int k0 = 0; k0 < K - 64; k0 += 64) {
    v16h A1 = load_frag_k(Ab, K, row0, k0 + 32, lane);
    v16h B1[4];
#pragma unroll
    for (int t = 0; t < 4; ++t)
      B1[t] = load_frag_k(W, K, colBase + t * 16, k0 + 32, lane);
#pragma unroll
    for (int t = 0; t < 4; ++t)
      acc[t] = __builtin_amdgcn_wmma_f32_16x16x32_f16(
          false, A0, false, B0[t], (short)0, acc[t], false, false);
    A0 = load_frag_k(Ab, K, row0, k0 + 64, lane);          // unconditional refill
#pragma unroll
    for (int t = 0; t < 4; ++t)
      B0[t] = load_frag_k(W, K, colBase + t * 16, k0 + 64, lane);
#pragma unroll
    for (int t = 0; t < 4; ++t)
      acc[t] = __builtin_amdgcn_wmma_f32_16x16x32_f16(
          false, A1, false, B1[t], (short)0, acc[t], false, false);
  }
  {  // peeled final double-step: K-64 (in buffer 0) and K-32
    v16h A1 = load_frag_k(Ab, K, row0, K - 32, lane);
    v16h B1[4];
#pragma unroll
    for (int t = 0; t < 4; ++t)
      B1[t] = load_frag_k(W, K, colBase + t * 16, K - 32, lane);
#pragma unroll
    for (int t = 0; t < 4; ++t)
      acc[t] = __builtin_amdgcn_wmma_f32_16x16x32_f16(
          false, A0, false, B0[t], (short)0, acc[t], false, false);
#pragma unroll
    for (int t = 0; t < 4; ++t)
      acc[t] = __builtin_amdgcn_wmma_f32_16x16x32_f16(
          false, A1, false, B1[t], (short)0, acc[t], false, false);
  }

  const int l = lane & 15, hi = lane >> 4;
  _Float16* Ob = Out + (size_t)batch * oBatchStride;
#pragma unroll
  for (int t = 0; t < 4; ++t) {
    int col = colBase + t * 16 + l;
    float bv = bias ? bias[col] : 0.0f;
#pragma unroll
    for (int r = 0; r < 8; ++r) {
      int row = row0 + r + hi * 8;   // C/D layout: lanes 0-15 -> M=r, 16-31 -> M=r+8
      Ob[(size_t)row * M + col] = (_Float16)(acc[t][r] + bv);
    }
  }
}

// ---------------------------------------------------------------------------
// Masked local attention: |dy|+|dx| <= 2 on the 64x32 grid => <= 13 neighbors.
// One thread per (bs, pixel, head); d = 64, scale = 1/8; ctx -> f16 (n, 256).
// ---------------------------------------------------------------------------
__global__ void local_attn(const _Float16* __restrict__ Q,
                           const _Float16* __restrict__ Kb,
                           const _Float16* __restrict__ Vb,
                           _Float16* __restrict__ Ctx) {
  int idx = blockIdx.x * blockDim.x + threadIdx.x;   // NBS*NPIX*4
  if (idx >= NBS * NPIX * 4) return;
  int h  = idx & 3;
  int p  = (idx >> 2) & (NPIX - 1);
  int bs = idx >> 13;
  int y = p >> 5, x = p & 31;

  const int dy[13] = {-2,-1,-1,-1, 0, 0, 0, 0, 0, 1, 1, 1, 2};
  const int dx[13] = { 0,-1, 0, 1,-2,-1, 0, 1, 2,-1, 0, 1, 0};

  const size_t base = (size_t)bs * NPIX * CDIM + h * HDIM;
  const _Float16* qv = Q + base + (size_t)p * CDIM;
  float qf[HDIM];
#pragma unroll
  for (int j = 0; j < HDIM; ++j) qf[j] = (float)qv[j];

  float sc[13];
  int   nbp[13];
  int   nn = 0;
  float mx = -1e30f;
#pragma unroll
  for (int i = 0; i < 13; ++i) {
    int yy = y + dy[i], xx = x + dx[i];
    if (yy < 0 || yy >= 64 || xx < 0 || xx >= 32) continue;
    int pn = yy * 32 + xx;
    const _Float16* kv = Kb + base + (size_t)pn * CDIM;
    float d = 0.0f;
#pragma unroll 8
    for (int j = 0; j < HDIM; ++j) d += qf[j] * (float)kv[j];
    d *= 0.125f;                 // 1/sqrt(64)
    sc[nn] = d; nbp[nn] = pn; ++nn;
    mx = fmaxf(mx, d);
  }
  float sum = 0.0f;
  for (int i = 0; i < nn; ++i) { sc[i] = __expf(sc[i] - mx); sum += sc[i]; }
  float inv = 1.0f / sum;

  float ctx[HDIM];
#pragma unroll
  for (int j = 0; j < HDIM; ++j) ctx[j] = 0.0f;
  for (int i = 0; i < nn; ++i) {
    const _Float16* vv = Vb + base + (size_t)nbp[i] * CDIM;
    float w = sc[i] * inv;
#pragma unroll 8
    for (int j = 0; j < HDIM; ++j) ctx[j] += w * (float)vv[j];
  }
  _Float16* cp = Ctx + base + (size_t)p * CDIM;
#pragma unroll
  for (int j = 0; j < HDIM; ++j) cp[j] = (_Float16)ctx[j];
}

// ---------------------------------------------------------------------------
// Fused: y = asym @ proj_w^T + proj_b ; LayerNorm(256) ; ReLU ; write (b,c,h,w).
// Phase 1: build the 16x1024 asym tile [qt | ctx | |qt-ctx| | qt*ctx] in LDS
//          (qt/ctx each read ONCE; all four segments computed branch-free).
// Phase 2: GEMM over K=1024 with A-fragments from LDS (ds_load, linear k0,
//          row pitch padded to 1032 halves -> 4-bank skew, conflict-free) and
//          B ping-ponged from global; straight-line body, last step peeled.
// Phase 3: LayerNorm + ReLU from an LDS tile, coalesced writes.
// ---------------------------------------------------------------------------
__global__ void proj_ln_relu(const _Float16* __restrict__ Xq,
                             const _Float16* __restrict__ Ctx,
                             const _Float16* __restrict__ PW,   // (256, 1024) f16
                             const float* __restrict__ pb,
                             const float* __restrict__ ln_g,
                             const float* __restrict__ ln_b,
                             float* __restrict__ out) {
  __shared__ __align__(16) _Float16 asym[16 * APITCH];   // 33 KB
  __shared__ float ly[16 * LNP];
  __shared__ float s_mean[16], s_istd[16];

  const int lane = threadIdx.x & 31;
  const int wave = threadIdx.x >> 5;
  const int bs = blockIdx.y;
  const int pBase = blockIdx.x * 16;
  const int l = lane & 15, hi = lane >> 4;

  // ---- Phase 1: build asym tile in LDS ----
  const _Float16* qBase = Xq  + ((size_t)bs * NPIX + pBase) * CDIM;
  const _Float16* cBase = Ctx + ((size_t)bs * NPIX + pBase) * CDIM;
  for (int i = threadIdx.x; i < 16 * CDIM; i += 128) {
    int row = i >> 8;            // 0..15
    int c   = i & 255;           // consecutive threads -> consecutive c
    float q = (float)qBase[(size_t)row * CDIM + c];
    float v = (float)cBase[(size_t)row * CDIM + c];
    _Float16* ap = asym + row * APITCH;
    ap[c]       = (_Float16)q;
    ap[256 + c] = (_Float16)v;
    ap[512 + c] = (_Float16)fabsf(q - v);
    ap[768 + c] = (_Float16)(q * v);
  }
  __syncthreads();

  // ---- Phase 2: GEMM, A from LDS, B ping-pong from global ----
  const _Float16* arow = asym;    // fragment rows are l-indexed inside loader
  v16h B0[4];
#pragma unroll
  for (int t = 0; t < 4; ++t)
    B0[t] = load_frag_k(PW, 4 * CDIM, wave * 64 + t * 16, 0, lane);

  v8f acc[4] = {};
#pragma unroll 1
  for (int k0 = 0; k0 < 4 * CDIM - 64; k0 += 64) {
    v16h B1[4];
#pragma unroll
    for (int t = 0; t < 4; ++t)
      B1[t] = load_frag_k(PW, 4 * CDIM, wave * 64 + t * 16, k0 + 32, lane);
    v16h a0 = load_frag_lds(arow, APITCH, k0, lane);
#pragma unroll
    for (int t = 0; t < 4; ++t)
      acc[t] = __builtin_amdgcn_wmma_f32_16x16x32_f16(
          false, a0, false, B0[t], (short)0, acc[t], false, false);
#pragma unroll
    for (int t = 0; t < 4; ++t)                              // unconditional refill
      B0[t] = load_frag_k(PW, 4 * CDIM, wave * 64 + t * 16, k0 + 64, lane);
    v16h a1 = load_frag_lds(arow, APITCH, k0 + 32, lane);
#pragma unroll
    for (int t = 0; t < 4; ++t)
      acc[t] = __builtin_amdgcn_wmma_f32_16x16x32_f16(
          false, a1, false, B1[t], (short)0, acc[t], false, false);
  }
  {  // peeled final double-step: k = 960 (buffer 0) and k = 992
    v16h B1[4];
#pragma unroll
    for (int t = 0; t < 4; ++t)
      B1[t] = load_frag_k(PW, 4 * CDIM, wave * 64 + t * 16, 4 * CDIM - 32, lane);
    v16h a0 = load_frag_lds(arow, APITCH, 4 * CDIM - 64, lane);
#pragma unroll
    for (int t = 0; t < 4; ++t)
      acc[t] = __builtin_amdgcn_wmma_f32_16x16x32_f16(
          false, a0, false, B0[t], (short)0, acc[t], false, false);
    v16h a1 = load_frag_lds(arow, APITCH, 4 * CDIM - 32, lane);
#pragma unroll
    for (int t = 0; t < 4; ++t)
      acc[t] = __builtin_amdgcn_wmma_f32_16x16x32_f16(
          false, a1, false, B1[t], (short)0, acc[t], false, false);
  }

  // ---- Phase 3: bias + LayerNorm + ReLU ----
#pragma unroll
  for (int t = 0; t < 4; ++t) {
    int col = wave * 64 + t * 16 + l;
    float bv = pb[col];
#pragma unroll
    for (int r = 0; r < 8; ++r) {
      int row = r + hi * 8;
      ly[row * LNP + col] = acc[t][r] + bv;
    }
  }
  __syncthreads();

  if (threadIdx.x < 16) {
    int r = threadIdx.x;
    float s = 0.0f;
    for (int ch = 0; ch < CDIM; ++ch) s += ly[r * LNP + ch];
    float mu = s * (1.0f / CDIM);
    float v = 0.0f;
    for (int ch = 0; ch < CDIM; ++ch) { float d = ly[r * LNP + ch] - mu; v += d * d; }
    s_mean[r] = mu;
    s_istd[r] = rsqrtf(v * (1.0f / CDIM) + 1e-5f);
  }
  __syncthreads();

  const int b = bs >> 1, s = bs & 1;
  for (int i = threadIdx.x; i < 16 * CDIM; i += 128) {
    int ch = i >> 4;
    int r  = i & 15;                 // consecutive threads -> consecutive x
    int p  = pBase + r;
    int y  = p >> 5, x = p & 31;
    float val = (ly[r * LNP + ch] - s_mean[r]) * s_istd[r] * ln_g[ch] + ln_b[ch];
    val = val > 0.0f ? val : 0.0f;
    out[(((size_t)b * CDIM + ch) * 64 + y) * 64 + s * 32 + x] = val;
  }
}

// ---------------------------------------------------------------------------
// Host-side orchestration
// ---------------------------------------------------------------------------
extern "C" void kernel_launch(void* const* d_in, const int* in_sizes, int n_in,
                              void* d_out, int out_size, void* d_ws, size_t ws_size,
                              hipStream_t stream) {
  (void)in_sizes; (void)n_in; (void)out_size; (void)ws_size;
  const float* feat   = (const float*)d_in[0];
  const float* in_w   = (const float*)d_in[1];
  const float* in_b   = (const float*)d_in[2];
  const float* out_w  = (const float*)d_in[3];
  const float* out_b  = (const float*)d_in[4];
  const float* proj_w = (const float*)d_in[5];
  const float* proj_b = (const float*)d_in[6];
  const float* ln_g   = (const float*)d_in[7];
  const float* ln_b   = (const float*)d_in[8];
  float* out = (float*)d_out;

  const size_t SLICE = (size_t)NBS * NPIX * CDIM;      // 8.39M halves per buffer
  char* ws = (char*)d_ws;
  size_t off = 0;
  auto alloc = [&](size_t bytes) { size_t r = off; off = (off + bytes + 255) & ~(size_t)255; return r; };
  _Float16* Xq    = (_Float16*)(ws + alloc(SLICE * 2));
  _Float16* Xm    = (_Float16*)(ws + alloc(SLICE * 2));   // later reused as ctxRaw
  _Float16* Wf    = (_Float16*)(ws + alloc((size_t)768 * 256 * 2));
  _Float16* OWf   = (_Float16*)(ws + alloc((size_t)256 * 256 * 2));
  _Float16* PWf   = (_Float16*)(ws + alloc((size_t)256 * 1024 * 2));
  _Float16* Qb    = (_Float16*)(ws + alloc(SLICE * 2));   // later reused as ctxProj
  _Float16* Kb    = (_Float16*)(ws + alloc(SLICE * 2));
  _Float16* Vb    = (_Float16*)(ws + alloc(SLICE * 2));

  // 1) weights -> f16
  f32_to_f16<<<(768 * 256 + 255) / 256, 256, 0, stream>>>(in_w,  Wf,  768 * 256);
  f32_to_f16<<<(256 * 256 + 255) / 256, 256, 0, stream>>>(out_w, OWf, 256 * 256);
  f32_to_f16<<<(256 * 1024 + 255) / 256, 256, 0, stream>>>(proj_w, PWf, 256 * 1024);

  // 2) build Xq / Xm
  {
    int total = NBS * CDIM * NPIX;
    prep_inputs<<<(total + 255) / 256, 256, 0, stream>>>(feat, Xq, Xm);
  }

  // 3) QKV GEMMs (batched over 16 slices): rows 2048, cols 256, K 256
  dim3 gg(NPIX / 64, CDIM / 64, NBS);
  size_t bstride = (size_t)NPIX * CDIM;
  gemm_xwt_f16<CDIM><<<gg, 128, 0, stream>>>(Xq, bstride, Wf,             in_b,       Qb, bstride, CDIM);
  gemm_xwt_f16<CDIM><<<gg, 128, 0, stream>>>(Xm, bstride, Wf + 256 * 256, in_b + 256, Kb, bstride, CDIM);
  gemm_xwt_f16<CDIM><<<gg, 128, 0, stream>>>(Xm, bstride, Wf + 512 * 256, in_b + 512, Vb, bstride, CDIM);

  // 4) sparse local attention; writes ctxRaw into Xm (Xm is dead now)
  {
    int total = NBS * NPIX * 4;
    local_attn<<<(total + 127) / 128, 128, 0, stream>>>(Qb, Kb, Vb, Xm);
  }

  // 5) out projection: ctxProj = ctxRaw @ out_w^T + out_b  (Qb is dead -> reuse)
  gemm_xwt_f16<CDIM><<<gg, 128, 0, stream>>>(Xm, bstride, OWf, out_b, Qb, bstride, CDIM);

  // 6) fused asym-proj + LayerNorm + ReLU -> output (b, c, h, w)
  {
    dim3 gf(NPIX / 16, NBS);
    proj_ln_relu<<<gf, 128, 0, stream>>>(Xq, Qb, PWf, proj_b, ln_g, ln_b, out);
  }
}
